// TaskTaskGAT2Layer_60318520705358
// MI455X (gfx1250) — matrix operand order repaired and checked by reading, not compiled, not run on hardware
//
#include <hip/hip_runtime.h>

// ---------------------------------------------------------------------------
// TaskTaskGAT2Layer for MI455X (gfx1250, wave32).
// Dense projections -> v_wmma_f32_16x16x32_f16, LDS tiles pre-swizzled so
// fragments load as contiguous b128 chunks. A tile shared by 8 waves/block.
// Graph softmax/aggregation -> wave-per-edge gather/scatter, f32 atomics.
// ---------------------------------------------------------------------------

#define NN 50000
#define EE 800000
#define DD 128

typedef __attribute__((ext_vector_type(16))) _Float16 v16h;
typedef __attribute__((ext_vector_type(4)))  _Float16 v4h;
typedef __attribute__((ext_vector_type(8)))  float    v8f;

__device__ __forceinline__ unsigned f2key(float f) {
    unsigned b = __float_as_uint(f);
    return (b & 0x80000000u) ? ~b : (b | 0x80000000u);
}
__device__ __forceinline__ float key2f(unsigned k) {
    return (k & 0x80000000u) ? __uint_as_float(k & 0x7FFFFFFFu)
                             : __uint_as_float(~k);
}

// ------------- weight convert + transpose: W[K,Ncol] f32 -> Wt[Ncol,K] f16 -
__global__ void wt_convert_kernel(const float* __restrict__ W,
                                  _Float16* __restrict__ Wt, int K, int Ncol) {
    int t = blockIdx.x * blockDim.x + threadIdx.x;
    if (t >= K * Ncol) return;
    int k = t / Ncol, n = t - k * Ncol;
    Wt[(size_t)n * K + k] = (_Float16)W[t];
}

// ---------------- zero init ------------------------------------------------
__global__ void zero_kernel(float* __restrict__ p, size_t n) {
    size_t t = (size_t)blockIdx.x * blockDim.x + threadIdx.x;
    if (t < n) p[t] = 0.0f;
}

// ---------------- WMMA GEMM: Y[N,Ncol] = X[N,K] @ W[K,Ncol] + bias ---------
// Block = Ncol/16 waves; wave w computes the 16x16 tile at columns w*16.
// A tile (16x32 f32 -> f16) staged once per block, pre-swizzled so each
// lane's fragment is 16 contiguous halfs (two ds_load_b128).
// Wt is the transposed f16 weight [Ncol,K] so B staging is contiguous b64.
__global__ __launch_bounds__(256)
void gemm_wmma_f16(const float* __restrict__ X, const _Float16* __restrict__ Wt,
                   const float* __restrict__ bias, float* __restrict__ Y,
                   int K, int Ncol) {
    __shared__ __attribute__((aligned(32))) _Float16 Ahs[512];       // 16x32
    __shared__ __attribute__((aligned(32))) _Float16 Bhs[8 * 512];   // per-wave 32x16
    const int tid  = threadIdx.x;
    const int lane = tid & 31;
    const int wid  = tid >> 5;
    const int m0   = blockIdx.x * 16;
    const int n0w  = wid * 16;
    const int half = lane >> 4;

    v8f acc = {};
    for (int k0 = 0; k0 < K; k0 += 32) {
        // ---- stage A (tid 0..127): 128 float4 chunks of the 16x32 tile ----
        if (tid < 128) {
            int r  = tid >> 3;          // row 0..15
            int c  = (tid & 7) * 4;     // k offset 0..28
            const float4 f4 = *(const float4*)&X[(size_t)(m0 + r) * K + k0 + c];
            int kb   = c & 15;
            int sel  = (kb >= 8) ? 1 : 0;
            int j0   = (kb - 8 * sel) + 8 * (c >> 4);
            int laneA = r + 16 * sel;
            v4h h4; h4[0] = (_Float16)f4.x; h4[1] = (_Float16)f4.y;
                    h4[2] = (_Float16)f4.z; h4[3] = (_Float16)f4.w;
            *(v4h*)&Ahs[laneA * 16 + j0] = h4;
        }
        // ---- stage B (per wave): 32x16 halfs from Wt rows n0w..n0w+15 -----
        #pragma unroll
        for (int t = 0; t < 4; ++t) {
            int idx = t * 32 + lane;    // 0..127 chunks of 4 halfs
            int r   = idx >> 3;         // n offset 0..15
            int k   = (idx & 7) * 4;    // k offset 0..28
            v4h h4 = *(const v4h*)&Wt[(size_t)(n0w + r) * K + k0 + k];
            int laneB = r + 16 * ((k >> 3) & 1);
            int j0    = (k & 7) + 8 * (k >> 4);
            *(v4h*)&Bhs[wid * 512 + laneB * 16 + j0] = h4;
        }
        __syncthreads();
        v16h a = *(const v16h*)&Ahs[lane * 16];
        v16h b = *(const v16h*)&Bhs[wid * 512 + lane * 16];
        acc = __builtin_amdgcn_wmma_f32_16x16x32_f16(
                  false, a, false, b, (short)0, acc, false, false);
        __syncthreads();
    }
    const int n  = n0w + (lane & 15);
    const float bv = bias[n];
    #pragma unroll
    for (int r = 0; r < 8; ++r) {
        int m = m0 + r + 8 * half;       // C/D layout: VGPR r -> M = r + 8*half
        Y[(size_t)m * Ncol + n] = acc[r] + bv;
    }
}

// ---------------- edge pass 1: alpha + segment max -------------------------
__global__ __launch_bounds__(256)
void edge_alpha_kernel(const float* __restrict__ xl, const float* __restrict__ xr,
                       const int* __restrict__ srcI, const int* __restrict__ dstI,
                       const float* __restrict__ att,
                       float* __restrict__ alpha, unsigned* __restrict__ amax,
                       int E) {
    int e    = (int)((blockIdx.x * (size_t)blockDim.x + threadIdx.x) >> 5);
    int lane = threadIdx.x & 31;
    if (e >= E) return;
    int s = srcI[e], d = dstI[e];
    const float* pj = xl + (size_t)s * 128;
    const float* pi = xr + (size_t)d * 128;
    #pragma unroll
    for (int h = 0; h < 2; ++h) {
        float p = 0.0f;
        #pragma unroll
        for (int t = 0; t < 2; ++t) {
            int c = lane + t * 32;
            float v = pi[h * 64 + c] + pj[h * 64 + c];
            v = v >= 0.0f ? v : 0.2f * v;              // LeakyReLU(0.2)
            p += att[h * 64 + c] * v;
        }
        #pragma unroll
        for (int off = 16; off > 0; off >>= 1)
            p += __shfl_down(p, off, 32);
        if (lane == 0) {
            alpha[(size_t)e * 2 + h] = p;
            atomicMax(&amax[(size_t)d * 2 + h], f2key(p));
        }
    }
}

// ---------------- edge pass 2: exp + segment sum ---------------------------
__global__ void edge_exp_den_kernel(float* __restrict__ alpha_ex,
                                    const unsigned* __restrict__ amax,
                                    const int* __restrict__ dstI,
                                    float* __restrict__ den, int E) {
    int t = blockIdx.x * blockDim.x + threadIdx.x;
    if (t >= E * 2) return;
    int e = t >> 1, h = t & 1;
    int d = dstI[e];
    float m  = key2f(amax[(size_t)d * 2 + h]);
    float ev = expf(alpha_ex[t] - m);
    alpha_ex[t] = ev;
    atomicAdd(&den[(size_t)d * 2 + h], ev);
}

// ---------------- edge pass 3: weighted scatter ----------------------------
__global__ __launch_bounds__(256)
void edge_scatter_kernel(const float* __restrict__ xl, const float* __restrict__ ex,
                         const float* __restrict__ den,
                         const int* __restrict__ srcI, const int* __restrict__ dstI,
                         float* __restrict__ accum, int E) {
    int e    = (int)((blockIdx.x * (size_t)blockDim.x + threadIdx.x) >> 5);
    int lane = threadIdx.x & 31;
    if (e >= E) return;
    int s = srcI[e], d = dstI[e];
    float a0 = ex[(size_t)e * 2 + 0] / den[(size_t)d * 2 + 0];
    float a1 = ex[(size_t)e * 2 + 1] / den[(size_t)d * 2 + 1];
    const float* pj = xl + (size_t)s * 128;
    float* po = accum + (size_t)d * 128;
    #pragma unroll
    for (int r = 0; r < 4; ++r) {
        int i = lane + r * 32;
        float aa = (i < 64) ? a0 : a1;
        atomicAdd(&po[i], pj[i] * aa);
    }
}

// ---------------- finalize: head-mean + residual + LN + lrelu(0.01) --------
__global__ __launch_bounds__(256)
void finalize_ln_kernel(const float* __restrict__ accum, const float* __restrict__ res,
                        const float* __restrict__ g, const float* __restrict__ b,
                        float* __restrict__ out, int n) {
    int node = (int)((blockIdx.x * (size_t)blockDim.x + threadIdx.x) >> 5);
    int lane = threadIdx.x & 31;
    if (node >= n) return;
    const float* pa = accum + (size_t)node * 128;
    const float* pr = res   + (size_t)node * 64;
    float v0 = 0.5f * (pa[lane]      + pa[64 + lane]) + pr[lane];
    float v1 = 0.5f * (pa[lane + 32] + pa[96 + lane]) + pr[lane + 32];
    float s = v0 + v1;
    #pragma unroll
    for (int off = 16; off > 0; off >>= 1) s += __shfl_xor(s, off, 32);
    float mean = s * (1.0f / 64.0f);
    float d0 = v0 - mean, d1 = v1 - mean;
    float q = d0 * d0 + d1 * d1;
    #pragma unroll
    for (int off = 16; off > 0; off >>= 1) q += __shfl_xor(q, off, 32);
    float inv = rsqrtf(q * (1.0f / 64.0f) + 1e-5f);
    float y0 = d0 * inv * g[lane]      + b[lane];
    float y1 = d1 * inv * g[lane + 32] + b[lane + 32];
    y0 = y0 >= 0.0f ? y0 : 0.01f * y0;
    y1 = y1 >= 0.0f ? y1 : 0.01f * y1;
    out[(size_t)node * 64 + lane]      = y0;
    out[(size_t)node * 64 + lane + 32] = y1;
}

// ---------------- finalize (layer 2): lrelu only, strided into d_out -------
__global__ void finalize_plain_kernel(const float* __restrict__ accum,
                                      const float* __restrict__ res,
                                      float* __restrict__ out, int colOff, int n) {
    int t = blockIdx.x * blockDim.x + threadIdx.x;
    if (t >= n * 64) return;
    int node = t >> 6, c = t & 63;
    float v = 0.5f * (accum[(size_t)node * 128 + c] +
                      accum[(size_t)node * 128 + 64 + c]) +
              res[(size_t)node * 64 + c];
    v = v >= 0.0f ? v : 0.01f * v;
    out[(size_t)node * 256 + colOff + c] = v;
}

// ---------------- copy x into out[:,128:256] -------------------------------
__global__ void copy_x_kernel(const float* __restrict__ x, float* __restrict__ out,
                              int n) {
    int t = blockIdx.x * blockDim.x + threadIdx.x;
    if (t >= n * 128) return;
    int node = t >> 7, c = t & 127;
    out[(size_t)node * 256 + 128 + c] = x[t];
}

// ---------------------------------------------------------------------------
extern "C" void kernel_launch(void* const* d_in, const int* in_sizes, int n_in,
                              void* d_out, int out_size, void* d_ws, size_t ws_size,
                              hipStream_t stream) {
    (void)in_sizes; (void)n_in; (void)out_size; (void)ws_size;
    const float* x    = (const float*)d_in[0];
    const int*   ei   = (const int*)d_in[1];
    const int*   srcA = ei;           // edge_index[0]
    const int*   dstA = ei + EE;      // edge_index[1]

    struct ConvP { const float *Wl, *bl, *Wr, *br, *att, *Wres, *bias; };
    ConvP cp[4];
    for (int i = 0; i < 4; ++i) {
        int base = 2 + i * 7;
        cp[i] = { (const float*)d_in[base + 0], (const float*)d_in[base + 1],
                  (const float*)d_in[base + 2], (const float*)d_in[base + 3],
                  (const float*)d_in[base + 4], (const float*)d_in[base + 5],
                  (const float*)d_in[base + 6] };
    }
    const float* nd_g = (const float*)d_in[30];
    const float* nd_b = (const float*)d_in[31];
    const float* nt_g = (const float*)d_in[32];
    const float* nt_b = (const float*)d_in[33];

    // ---- workspace layout (f32 unless noted) ----
    float* ws    = (float*)d_ws;
    float* xl    = ws;                               // N*128
    float* xr    = xl    + (size_t)NN * 128;         // N*128
    float* res   = xr    + (size_t)NN * 128;         // N*64
    float* accum = res   + (size_t)NN * 64;          // N*128  } zeroed together
    float* den   = accum + (size_t)NN * 128;         // N*2    }
    unsigned* amax = (unsigned*)(den + (size_t)NN * 2); // N*2  }
    float* ex    = (float*)(amax + (size_t)NN * 2);  // E*2
    float* dep   = ex    + (size_t)EE * 2;           // N*64
    float* dant  = dep   + (size_t)NN * 64;          // N*64
    _Float16* wh = (_Float16*)(dant + (size_t)NN * 64);

    // f16 transposed weight slots [Ncol,K]
    const int Ks[4] = {128, 128, 64, 64};
    _Float16 *whl[4], *whr[4], *whres[4];
    {
        _Float16* p = wh;
        for (int i = 0; i < 4; ++i) {
            whl[i]   = p; p += (size_t)Ks[i] * 128;
            whr[i]   = p; p += (size_t)Ks[i] * 128;
            whres[i] = p; p += (size_t)Ks[i] * 64;
        }
    }
    for (int i = 0; i < 4; ++i) {
        int nWl = Ks[i] * 128, nWres = Ks[i] * 64;
        wt_convert_kernel<<<(nWl + 255) / 256, 256, 0, stream>>>(cp[i].Wl,   whl[i],   Ks[i], 128);
        wt_convert_kernel<<<(nWl + 255) / 256, 256, 0, stream>>>(cp[i].Wr,   whr[i],   Ks[i], 128);
        wt_convert_kernel<<<(nWres + 255) / 256, 256, 0, stream>>>(cp[i].Wres, whres[i], Ks[i], 64);
    }

    auto run_conv = [&](const float* feat, int K, int ci,
                        const int* sI, const int* dI) {
        // GEMMs: block = Ncol/16 waves, grid = row tiles
        gemm_wmma_f16<<<NN / 16, 256, 0, stream>>>(feat, whl[ci],   cp[ci].bl,   xl,  K, 128);
        gemm_wmma_f16<<<NN / 16, 256, 0, stream>>>(feat, whr[ci],   cp[ci].br,   xr,  K, 128);
        gemm_wmma_f16<<<NN / 16, 128, 0, stream>>>(feat, whres[ci], cp[ci].bias, res, K, 64);
        size_t zn = (size_t)NN * 132;  // accum + den + amax (contiguous)
        zero_kernel<<<(unsigned)((zn + 255) / 256), 256, 0, stream>>>(accum, zn);
        edge_alpha_kernel<<<(EE + 7) / 8, 256, 0, stream>>>(xl, xr, sI, dI,
                                                            cp[ci].att, ex, amax, EE);
        edge_exp_den_kernel<<<(2 * EE + 255) / 256, 256, 0, stream>>>(ex, amax, dI, den, EE);
        edge_scatter_kernel<<<(EE + 7) / 8, 256, 0, stream>>>(xl, ex, den, sI, dI, accum, EE);
    };

    float* out = (float*)d_out;

    // Layer 1, dep direction (src -> dst), then LN(nd) + lrelu(0.01)
    run_conv(x, 128, 0, srcA, dstA);
    finalize_ln_kernel<<<(NN + 7) / 8, 256, 0, stream>>>(accum, res, nd_g, nd_b, dep, NN);

    // Layer 1, dant direction (dst -> src), then LN(nt) + lrelu(0.01)
    run_conv(x, 128, 1, dstA, srcA);
    finalize_ln_kernel<<<(NN + 7) / 8, 256, 0, stream>>>(accum, res, nt_g, nt_b, dant, NN);

    // Layer 2, dep2 -> out[:, 0:64]
    run_conv(dep, 64, 2, srcA, dstA);
    finalize_plain_kernel<<<(NN * 64 + 255) / 256, 256, 0, stream>>>(accum, res, out, 0, NN);

    // Layer 2, dant2 -> out[:, 64:128]
    run_conv(dant, 64, 3, dstA, srcA);
    finalize_plain_kernel<<<(NN * 64 + 255) / 256, 256, 0, stream>>>(accum, res, out, 64, NN);

    // x -> out[:, 128:256]
    copy_x_kernel<<<(NN * 128 + 255) / 256, 256, 0, stream>>>(x, out, NN);
}
